// ContinuousMambaBlock_71236327571723
// MI455X (gfx1250) — matrix operand order repaired and checked by reading, compiled, not run on hardware
//
#include <hip/hip_runtime.h>
#include <hip/hip_bf16.h>
#include <math.h>

// Problem constants (from reference)
constexpr int Bn  = 8;
constexpr int Sn  = 2048;
constexpr int Dm  = 512;
constexpr int DSn = 16;
constexpr int Ln  = 2;
constexpr int DF  = 2048;
constexpr int Mr  = Bn * Sn;   // 16384 rows

typedef __bf16 bf16;
typedef bf16  bf16x16 __attribute__((ext_vector_type(16)));
typedef float f32x8   __attribute__((ext_vector_type(8)));

union FragAB { bf16x16 v; uint4 q[2]; };

// ---------------------------------------------------------------------------
// bf16 WMMA GEMM:  out[M,N] = A[M,K](bf16,row-major) * Bt[N,K]^T (bf16) + bias
// EPI: 0 = bias -> f32      (xproj)
//      1 = bias + aux(f32) -> f32            (oproj + residual add)
//      2 = bias, exact GELU -> bf16          (ffn1)
//      3 = bias + aux(f32) -> f32 AND bf16   (ffn2 + normed add, new x)
// Wave tile: 16 rows x (NT*16) cols. Block = 8 waves = 128 rows.
// Weights stay resident in the 192MB L2 -> no LDS staging needed.
// ---------------------------------------------------------------------------
template <int EPI, int NT>
__global__ __launch_bounds__(256) void gemm_bf16(
    const bf16* __restrict__ A, const bf16* __restrict__ Bt,
    const float* __restrict__ bias, const float* __restrict__ aux,
    float* __restrict__ outF, bf16* __restrict__ outB,
    int M, int N, int K)
{
  const int lane = threadIdx.x & 31;
  const int wave = threadIdx.x >> 5;
  const int row0 = blockIdx.y * 128 + wave * 16;
  const int col0 = blockIdx.x * (NT * 16);
  const int r    = lane & 15;   // A-row / B-col / D-col within tile
  const int half = lane >> 4;   // K-half selector

  f32x8 acc[NT];
#pragma unroll
  for (int t = 0; t < NT; ++t)
#pragma unroll
    for (int i = 0; i < 8; ++i) acc[t][i] = 0.f;

  const bf16* aRow = A + (size_t)(row0 + r) * K;
  for (int k0 = 0; k0 < K; k0 += 32) {
    // A 16x32 bf16 fragment: elems 0..7 = K(h*8..), elems 8..15 = K(16+h*8..)
    FragAB af;
    af.q[0] = *(const uint4*)(aRow + k0 + half * 8);
    af.q[1] = *(const uint4*)(aRow + k0 + 16 + half * 8);
#pragma unroll
    for (int t = 0; t < NT; ++t) {
      // B 32x16 fragment from transposed weight: elems 0..7 = K(h*8..), elems 8..15 = K(16+h*8..)
      const bf16* bRow = Bt + (size_t)(col0 + t * 16 + r) * K + k0 + half * 8;
      FragAB bfrag;
      bfrag.q[0] = *(const uint4*)(bRow);
      bfrag.q[1] = *(const uint4*)(bRow + 16);
      acc[t] = __builtin_amdgcn_wmma_f32_16x16x32_bf16(
          false, af.v, false, bfrag.v, (short)0, acc[t], false, false);
    }
  }

  // D layout: lane<16 -> rows row0+0..7, col r ; lane>=16 -> rows row0+8..15
  const int rbase = row0 + half * 8;
#pragma unroll
  for (int t = 0; t < NT; ++t) {
    const int c  = col0 + t * 16 + r;
    const float bv = bias[c];
#pragma unroll
    for (int i = 0; i < 8; ++i) {
      const size_t idx = (size_t)(rbase + i) * N + c;
      float v = acc[t][i] + bv;
      if constexpr (EPI == 1 || EPI == 3) v += aux[idx];
      if constexpr (EPI == 2) v = 0.5f * v * (1.f + erff(v * 0.70710678118654752f));
      if constexpr (EPI == 0 || EPI == 1 || EPI == 3) outF[idx] = v;
      if constexpr (EPI == 2 || EPI == 3) outB[idx] = (bf16)v;
    }
  }
}

// Convert weight [L,K,N] f32 -> transposed bf16 [L,N,K] (coalesced writes)
__global__ void k_convertT(const float* __restrict__ W, bf16* __restrict__ Wt,
                           int K, int N, int total)
{
  int id = blockIdx.x * blockDim.x + threadIdx.x;
  if (id >= total) return;
  int l = id / (N * K);
  int rem = id - l * N * K;
  int n = rem / K;
  int k = rem - n * K;
  Wt[id] = (bf16)W[(size_t)l * K * N + (size_t)k * N + n];
}

// v = kmote_W @ fkan_W @ dproj_W ; c = fkan_b @ dproj_W + dproj_b  (rank-1 collapse)
__global__ void k_prep1(const float* __restrict__ kW, const float* __restrict__ fkanW,
                        const float* __restrict__ fkanB, const float* __restrict__ dprojW,
                        const float* __restrict__ dprojB,
                        float* __restrict__ v, float* __restrict__ c)
{
  __shared__ float tsh[Dm];
  int j = threadIdx.x;
  float acc = 0.f;
  for (int i = 0; i < Dm; ++i) acc += kW[i] * fkanW[i * Dm + j];
  tsh[j] = acc;
  __syncthreads();
  float vv = 0.f, cc = 0.f;
  for (int i = 0; i < Dm; ++i) {
    float w = dprojW[i * Dm + j];
    vv += tsh[i] * w;
    cc += fkanB[i] * w;
  }
  v[j] = vv;
  c[j] = cc + dprojB[j];
}

// Per-layer collapsed dp1/dp2 vectors:
// w1=v@dp1, cb1=c@dp1+b1, w2bar[k]=mean_j dp2[k,j], b2bar=mean(b2)
__global__ void k_prep2(const float* __restrict__ v, const float* __restrict__ c,
                        const float* __restrict__ dp1W, const float* __restrict__ dp1b,
                        const float* __restrict__ dp2W, const float* __restrict__ dp2b,
                        float* __restrict__ w1, float* __restrict__ cb1,
                        float* __restrict__ w2bar, float* __restrict__ b2bar)
{
  int l = blockIdx.x, k = threadIdx.x;
  const float* W1 = dp1W + (size_t)l * Dm * Dm;
  float a = 0.f, b = 0.f;
  for (int i = 0; i < Dm; ++i) { float w = W1[i * Dm + k]; a += v[i] * w; b += c[i] * w; }
  w1[l * Dm + k]  = a;
  cb1[l * Dm + k] = b + dp1b[l * Dm + k];
  const float* W2 = dp2W + (size_t)l * Dm * Dm + (size_t)k * Dm;
  float s = 0.f;
  for (int j = 0; j < Dm; ++j) s += W2[j];
  w2bar[l * Dm + k] = s * (1.f / Dm);
  if (k == 0) {
    float t = 0.f;
    for (int j = 0; j < Dm; ++j) t += dp2b[l * Dm + j];
    b2bar[l] = t * (1.f / Dm);
  }
}

// x = ts*kmote_W + kmote_b (f32 + bf16) ; delta[m] = ts - prev_ts
__global__ void k_embed(const float* __restrict__ T, const float* __restrict__ kW,
                        const float* __restrict__ kb,
                        float* __restrict__ xF, bf16* __restrict__ xB,
                        float* __restrict__ delta)
{
  int id = blockIdx.x * blockDim.x + threadIdx.x;
  int m = id >> 9, d = id & 511;
  float t = T[m];
  float val = t * kW[d] + kb[d];
  xF[id] = val;
  xB[id] = (bf16)val;
  if (d == 0) {
    int s = m & (Sn - 1);
    float prev = (s == 0) ? t : T[m - 1];
    delta[m] = t - prev;
  }
}

// dmean[m] = sum_k softplus(delta*w1+cb1)*w2bar + b2bar ; xmean[m] = mean(x row)
__global__ __launch_bounds__(256) void k_rowstats(
    const float* __restrict__ xF, const float* __restrict__ delta,
    const float* __restrict__ w1, const float* __restrict__ cb1,
    const float* __restrict__ w2bar, const float* __restrict__ b2bar,
    float* __restrict__ dmean, float* __restrict__ xmean, int l)
{
  int wave = threadIdx.x >> 5, lane = threadIdx.x & 31;
  int m = blockIdx.x * 8 + wave;
  const float* w1l = w1 + l * Dm;
  const float* c1l = cb1 + l * Dm;
  const float* w2l = w2bar + l * Dm;
  float dlt = delta[m];
  float sd = 0.f, sx = 0.f;
  for (int k = lane; k < Dm; k += 32) {
    float z = dlt * w1l[k] + c1l[k];
    float sp = (z > 20.f) ? z : log1pf(expf(z));
    sd += sp * w2l[k];
    sx += xF[(size_t)m * Dm + k];
  }
  for (int off = 16; off; off >>= 1) {
    sd += __shfl_xor(sd, off, 32);
    sx += __shfl_xor(sx, off, 32);
  }
  if (lane == 0) { dmean[m] = sd + b2bar[l]; xmean[m] = sx * (1.f / Dm); }
}

// a = 1 + dmean*(-exp(A_log)) ; u = dmean * Bm * xmean
__global__ void k_ssm_prep(const float* __restrict__ dmean, const float* __restrict__ xmean,
                           const float* __restrict__ BC, const float* __restrict__ A_log,
                           float* __restrict__ aArr, float* __restrict__ uArr, int l)
{
  int id = blockIdx.x * blockDim.x + threadIdx.x;   // M*DSn
  int m = id >> 4, n = id & 15;
  float dm = dmean[m];
  float A = -expf(A_log[l * DSn + n]);
  aArr[id] = 1.f + dm * A;
  uArr[id] = dm * BC[m * 32 + n] * xmean[m];
}

// 128 independent first-order recurrences h = a*h + u (loads independent of carry)
__global__ void k_scan(const float* __restrict__ aArr, const float* __restrict__ uArr,
                       float* __restrict__ hs)
{
  int tid = blockIdx.x * blockDim.x + threadIdx.x;  // B*DSn = 128
  int b = tid >> 4, n = tid & 15;
  float h = 0.f;
  size_t base = (size_t)b * Sn * DSn + n;
  for (int s = 0; s < Sn; ++s) {
    size_t idx = base + (size_t)s * DSn;
    h = aArr[idx] * h + uArr[idx];
    hs[idx] = h;
  }
}

// y[m,d] = dot16(hs[m],Cm[m]) + Dp[d]*x[m,d]  -> bf16 (oproj input)
__global__ __launch_bounds__(256) void k_y(
    const float* __restrict__ hs, const float* __restrict__ BC,
    const float* __restrict__ Dp, const float* __restrict__ xF,
    bf16* __restrict__ yB, int l)
{
  int wave = threadIdx.x >> 5, lane = threadIdx.x & 31;
  int m = blockIdx.x * 8 + wave;
  float p = (lane < DSn) ? hs[(size_t)m * DSn + lane] * BC[m * 32 + DSn + lane] : 0.f;
  for (int off = 16; off; off >>= 1) p += __shfl_xor(p, off, 32);
  const float* dpl = Dp + l * Dm;
  for (int d = lane; d < Dm; d += 32)
    yB[(size_t)m * Dm + d] = (bf16)(p + dpl[d] * xF[(size_t)m * Dm + d]);
}

// LayerNorm over the (ssm_out + x) residual -> normed f32 + bf16
__global__ __launch_bounds__(256) void k_layernorm(
    const float* __restrict__ resid, const float* __restrict__ g,
    const float* __restrict__ b, float* __restrict__ nF, bf16* __restrict__ nB, int l)
{
  int wave = threadIdx.x >> 5, lane = threadIdx.x & 31;
  int m = blockIdx.x * 8 + wave;
  const float* row = resid + (size_t)m * Dm;
  float vals[16];
  float s = 0.f;
#pragma unroll
  for (int i = 0; i < 16; ++i) { vals[i] = row[lane + i * 32]; s += vals[i]; }
  for (int off = 16; off; off >>= 1) s += __shfl_xor(s, off, 32);
  float mu = s * (1.f / Dm);
  float q = 0.f;
#pragma unroll
  for (int i = 0; i < 16; ++i) { float d = vals[i] - mu; q += d * d; }
  for (int off = 16; off; off >>= 1) q += __shfl_xor(q, off, 32);
  float rinv = rsqrtf(q * (1.f / Dm) + 1e-5f);
  const float* gl = g + l * Dm;
  const float* bl = b + l * Dm;
#pragma unroll
  for (int i = 0; i < 16; ++i) {
    int d = lane + i * 32;
    float v = (vals[i] - mu) * rinv * gl[d] + bl[d];
    nF[(size_t)m * Dm + d] = v;
    nB[(size_t)m * Dm + d] = (bf16)v;
  }
}

extern "C" void kernel_launch(void* const* d_in, const int* in_sizes, int n_in,
                              void* d_out, int out_size, void* d_ws, size_t ws_size,
                              hipStream_t stream)
{
  const float* ts      = (const float*)d_in[0];
  const float* kmote_W = (const float*)d_in[1];
  const float* kmote_b = (const float*)d_in[2];
  const float* fkan_W  = (const float*)d_in[3];
  const float* fkan_b  = (const float*)d_in[4];
  const float* dproj_W = (const float*)d_in[5];
  const float* dproj_b = (const float*)d_in[6];
  const float* A_log   = (const float*)d_in[7];
  const float* Dp      = (const float*)d_in[8];
  const float* xproj_W = (const float*)d_in[9];
  const float* xproj_b = (const float*)d_in[10];
  const float* dp1_W   = (const float*)d_in[11];
  const float* dp1_b   = (const float*)d_in[12];
  const float* dp2_W   = (const float*)d_in[13];
  const float* dp2_b   = (const float*)d_in[14];
  const float* oproj_W = (const float*)d_in[15];
  const float* oproj_b = (const float*)d_in[16];
  const float* ln_g    = (const float*)d_in[17];
  const float* ln_b    = (const float*)d_in[18];
  const float* ffn1_W  = (const float*)d_in[19];
  const float* ffn1_b  = (const float*)d_in[20];
  const float* ffn2_W  = (const float*)d_in[21];
  const float* ffn2_b  = (const float*)d_in[22];

  char* wp = (char*)d_ws;
  auto alloc = [&](size_t bytes) {
    void* p = (void*)wp;
    wp += (bytes + 255) & ~(size_t)255;
    return p;
  };

  float* xF     = (float*)alloc((size_t)Mr * Dm * 4);
  bf16*  xB     = (bf16*) alloc((size_t)Mr * Dm * 2);
  float* residF = (float*)alloc((size_t)Mr * Dm * 4);
  float* nF     = (float*)alloc((size_t)Mr * Dm * 4);
  bf16*  nB     = (bf16*) alloc((size_t)Mr * Dm * 2);
  bf16*  hB     = (bf16*) alloc((size_t)Mr * DF * 2);
  bf16*  yB     = (bf16*) alloc((size_t)Mr * Dm * 2);
  float* BC     = (float*)alloc((size_t)Mr * 32 * 4);
  float* aArr   = (float*)alloc((size_t)Mr * DSn * 4);
  float* uArr   = (float*)alloc((size_t)Mr * DSn * 4);
  float* hsArr  = (float*)alloc((size_t)Mr * DSn * 4);
  float* delta  = (float*)alloc((size_t)Mr * 4);
  float* dmean  = (float*)alloc((size_t)Mr * 4);
  float* xmean  = (float*)alloc((size_t)Mr * 4);
  float* vvec   = (float*)alloc(Dm * 4);
  float* cvec   = (float*)alloc(Dm * 4);
  float* w1     = (float*)alloc((size_t)Ln * Dm * 4);
  float* cb1    = (float*)alloc((size_t)Ln * Dm * 4);
  float* w2bar  = (float*)alloc((size_t)Ln * Dm * 4);
  float* b2bar  = (float*)alloc(Ln * 4);
  bf16* xprojT  = (bf16*)alloc((size_t)Ln * 32 * Dm * 2);
  bf16* oprojT  = (bf16*)alloc((size_t)Ln * Dm * Dm * 2);
  bf16* ffn1T   = (bf16*)alloc((size_t)Ln * DF * Dm * 2);
  bf16* ffn2T   = (bf16*)alloc((size_t)Ln * Dm * DF * 2);

  // Weight convert+transpose (one-shot; weights then stay hot in 192MB L2)
  {
    int t1 = Ln * 32 * Dm;
    k_convertT<<<(t1 + 255) / 256, 256, 0, stream>>>(xproj_W, xprojT, Dm, 32, t1);
    int t2 = Ln * Dm * Dm;
    k_convertT<<<(t2 + 255) / 256, 256, 0, stream>>>(oproj_W, oprojT, Dm, Dm, t2);
    int t3 = Ln * Dm * DF;
    k_convertT<<<(t3 + 255) / 256, 256, 0, stream>>>(ffn1_W, ffn1T, Dm, DF, t3);
    int t4 = Ln * DF * Dm;
    k_convertT<<<(t4 + 255) / 256, 256, 0, stream>>>(ffn2_W, ffn2T, DF, Dm, t4);
  }

  k_prep1<<<1, Dm, 0, stream>>>(kmote_W, fkan_W, fkan_b, dproj_W, dproj_b, vvec, cvec);
  k_prep2<<<Ln, Dm, 0, stream>>>(vvec, cvec, dp1_W, dp1_b, dp2_W, dp2_b, w1, cb1, w2bar, b2bar);
  k_embed<<<Mr * Dm / 256, 256, 0, stream>>>(ts, kmote_W, kmote_b, xF, xB, delta);

  for (int l = 0; l < Ln; ++l) {
    k_rowstats<<<Mr / 8, 256, 0, stream>>>(xF, delta, w1, cb1, w2bar, b2bar, dmean, xmean, l);

    // BC = x @ xproj + b   (N=32 -> 2 column tiles per wave)
    gemm_bf16<0, 2><<<dim3(1, Mr / 128), 256, 0, stream>>>(
        xB, xprojT + (size_t)l * 32 * Dm, xproj_b + l * 32, nullptr,
        BC, nullptr, Mr, 32, Dm);

    k_ssm_prep<<<Mr * DSn / 256, 256, 0, stream>>>(dmean, xmean, BC, A_log, aArr, uArr, l);
    k_scan<<<4, 32, 0, stream>>>(aArr, uArr, hsArr);
    k_y<<<Mr / 8, 256, 0, stream>>>(hsArr, BC, Dp, xF, yB, l);

    // resid = y @ oproj + b + x
    gemm_bf16<1, 8><<<dim3(Dm / 128, Mr / 128), 256, 0, stream>>>(
        yB, oprojT + (size_t)l * Dm * Dm, oproj_b + l * Dm, xF,
        residF, nullptr, Mr, Dm, Dm);

    k_layernorm<<<Mr / 8, 256, 0, stream>>>(residF, ln_g, ln_b, nF, nB, l);

    // h = gelu(normed @ ffn1 + b)  -> bf16
    gemm_bf16<2, 8><<<dim3(DF / 128, Mr / 128), 256, 0, stream>>>(
        nB, ffn1T + (size_t)l * DF * Dm, ffn1_b + l * DF, nullptr,
        nullptr, hB, Mr, DF, Dm);

    // x_new = h @ ffn2 + b + normed  (last layer writes straight to d_out)
    float* outx = (l == Ln - 1) ? (float*)d_out : xF;
    gemm_bf16<3, 8><<<dim3(Dm / 128, Mr / 128), 256, 0, stream>>>(
        hB, ffn2T + (size_t)l * Dm * DF, ffn2_b + l * Dm, nF,
        outx, xB, Mr, Dm, DF);
  }
}
